// SAGE_4028679323669
// MI455X (gfx1250) — matrix operand order repaired and checked by reading, not compiled, not run on hardware
//
#include <hip/hip_runtime.h>
#include <cstdint>
#include <cstddef>

// ---------------------------------------------------------------------------
// Types for CDNA5 WMMA (wave32): v_wmma_f32_16x16x32_bf16
// ---------------------------------------------------------------------------
typedef __attribute__((ext_vector_type(16))) __bf16 v16bf;
typedef __attribute__((ext_vector_type(8)))  float  v8f;
typedef __attribute__((ext_vector_type(4)))  unsigned int u32x4;
typedef __attribute__((ext_vector_type(2)))  unsigned int u32x2;

union BFrag { v16bf v; u32x4 q[2]; };
union BFCvt { unsigned short u; __bf16 b; };

__device__ inline __bf16 f2bf(float f) {
  unsigned int u = __float_as_uint(f);
  u += 0x7FFFu + ((u >> 16) & 1u);        // round-to-nearest-even
  BFCvt c; c.u = (unsigned short)(u >> 16);
  return c.b;
}

__device__ inline unsigned int pack2bf(float lo, float hi) {
  unsigned int ul = __float_as_uint(lo);
  ul += 0x7FFFu + ((ul >> 16) & 1u);
  unsigned int uh = __float_as_uint(hi);
  uh += 0x7FFFu + ((uh >> 16) & 1u);
  return (ul >> 16) | (uh & 0xFFFF0000u);
}

// A fragment (16x32 bf16, M x K) from row-major LDS [16][strideH]:
// lanes 0-15 hold M=lane,    K = kt*32 + {0..7, 16..23}
// lanes 16-31 hold M=lane-16, K = kt*32 + {8..15, 24..31}
__device__ inline v16bf load_a_frag(const __bf16* base, int strideH, int kt, int lane) {
  int m = lane & 15;
  int half = lane >> 4;
  const __bf16* p = base + m * strideH + kt * 32 + half * 8;
  BFrag f;
  f.q[0] = *(const u32x4*)p;
  f.q[1] = *(const u32x4*)(p + 16);
  return f.v;
}

// B fragment (32x16 bf16, K x N) from pre-packed global weights: each lane's
// 16 halfs stored contiguously (32B) -> two global b128 loads per lane.
__device__ inline v16bf load_b_frag(const __bf16* pk, int KT, int nt, int kt, int lane) {
  const __bf16* p = pk + ((size_t)(nt * KT + kt) * 32 + (size_t)lane) * 16;
  BFrag f;
  f.q[0] = *(const u32x4*)p;
  f.q[1] = *(const u32x4*)(p + 8);
  return f.v;
}

__device__ inline v8f wmma_bf16(v16bf a, v16bf b, v8f c) {
  return __builtin_amdgcn_wmma_f32_16x16x32_bf16(false, a, false, b, (short)0, c, false, false);
}

__device__ inline float sigm(float x) { return 1.0f / (1.0f + __expf(-x)); }

// ---------------------------------------------------------------------------
// One-time weight pack: W [R rows, C cols] f32 -> bf16 B-fragments of W^T.
// B[k][n] = W[n][k], K = C, Ncols = R. One thread produces one lane's 32B.
// ---------------------------------------------------------------------------
__global__ void pack_weights(const float* __restrict__ src, __bf16* __restrict__ dst,
                             int R, int C) {
  int KT = C >> 5;
  int total = (R >> 4) * KT * 32;
  for (int id = blockIdx.x * blockDim.x + threadIdx.x; id < total;
       id += gridDim.x * blockDim.x) {
    int lane = id & 31;
    int blk  = id >> 5;
    int kt = blk % KT;
    int nt = blk / KT;
    int n = (nt << 4) + (lane & 15);
    int kbase = (lane >> 4) << 3;
    __bf16* out = dst + (size_t)id * 16;
#pragma unroll
    for (int v = 0; v < 8; ++v) {
#pragma unroll
      for (int j = 0; j < 2; ++j) {
        int k = (kt << 5) + kbase + ((v >> 2) << 4) + ((v & 3) << 1) + j;
        out[v * 2 + j] = f2bf(src[(size_t)n * C + k]);
      }
    }
  }
}

// ---------------------------------------------------------------------------
// One SAGE layer: 16-node tile per block, 8 waves (256 thr, wave32).
// Gate n-tiles are assigned so each wave owns the SAME column slab for all
// four gates (i,f,g,o) -> LSTM cell state lives in registers; only h round-
// trips through LDS (bf16, as next step's WMMA A operand).
//   F=128/256, OUTF=256/64, FIRST: f32-feat in + relu/bf16 out vs bf16 in +
//   f32 out.
// ---------------------------------------------------------------------------
template <int F, int OUTF, bool FIRST>
__global__ __launch_bounds__(256) void sage_layer(
    const float* __restrict__ xF32, const __bf16* __restrict__ xBF,
    const int* __restrict__ nbr, const int* __restrict__ deg,
    const float* __restrict__ bih, const float* __restrict__ bhh,
    const float* __restrict__ bout,
    const __bf16* __restrict__ pWih, const __bf16* __restrict__ pWhh,
    const __bf16* __restrict__ pWself, const __bf16* __restrict__ pWneigh,
    __bf16* __restrict__ outBF, float* __restrict__ outF32) {
  constexpr int KT  = F / 32;     // WMMA K-steps
  constexpr int CPG = F / 16;     // column tiles per gate
  constexpr int TW  = CPG / 8;    // column tiles per wave per gate (1 or 2)

  __shared__ __bf16 xA[16 * F];   // gathered x_t (A operand)
  __shared__ __bf16 hA[16 * F];   // hidden state (A operand)

  const int tid  = threadIdx.x;
  const int lane = tid & 31;
  const int wave = tid >> 5;
  const int nodeBase = blockIdx.x << 4;
  const int m0 = (lane >> 4) * 8;        // accumulator row base for this lane
  const int colBase = wave * 16 * TW + (lane & 15);

  // Per-lane persistent LSTM state & constants (register resident).
  v8f cReg[TW], hReg[TW];
  float bI[TW], bF[TW], bG[TW], bO[TW];
  int degR[8];
#pragma unroll
  for (int j = 0; j < TW; ++j) {
    int col = colBase + j * 16;
    bI[j] = bih[col]         + bhh[col];
    bF[j] = bih[F + col]     + bhh[F + col];
    bG[j] = bih[2 * F + col] + bhh[2 * F + col];
    bO[j] = bih[3 * F + col] + bhh[3 * F + col];
#pragma unroll
    for (int r = 0; r < 8; ++r) { cReg[j][r] = 0.0f; hReg[j][r] = 0.0f; }
  }
#pragma unroll
  for (int r = 0; r < 8; ++r) degR[r] = deg[nodeBase + m0 + r];

  for (int t = 0; t < 16; ++t) {
    // ---- phase A: publish h (bf16) to LDS, gather x_t into LDS ----------
#pragma unroll
    for (int j = 0; j < TW; ++j)
#pragma unroll
      for (int r = 0; r < 8; ++r)
        hA[(m0 + r) * F + colBase + j * 16] = f2bf(hReg[j][r]);

    if (FIRST) {
      // 512 float4 chunks: 2 per thread.
#pragma unroll
      for (int it = 0; it < 2; ++it) {
        int e = tid + it * 256;
        int r = e / (F / 4), c4 = e % (F / 4);
        int idx = nbr[(size_t)(nodeBase + r) * 16 + t];
        const float4 v = *(const float4*)(xF32 + (size_t)idx * F + c4 * 4);
        u32x2 p; p.x = pack2bf(v.x, v.y); p.y = pack2bf(v.z, v.w);
        *(u32x2*)(xA + r * F + c4 * 4) = p;
      }
    } else {
      // 512 8-half chunks: 2 per thread.
#pragma unroll
      for (int it = 0; it < 2; ++it) {
        int e = tid + it * 256;
        int r = e / (F / 8), c8 = e % (F / 8);
        int idx = nbr[(size_t)(nodeBase + r) * 16 + t];
        *(u32x4*)(xA + r * F + c8 * 8) =
            *(const u32x4*)(xBF + (size_t)idx * F + c8 * 8);
      }
    }
    __syncthreads();

    // ---- phase B: gates = x_t @ Wih^T + h @ Whh^T (WMMA) ---------------
    v8f acc[4][TW];
#pragma unroll
    for (int g = 0; g < 4; ++g)
#pragma unroll
      for (int j = 0; j < TW; ++j)
#pragma unroll
        for (int r = 0; r < 8; ++r) acc[g][j][r] = 0.0f;

#pragma unroll
    for (int kt = 0; kt < KT; ++kt) {
      v16bf aX = load_a_frag(xA, F, kt, lane);
      v16bf aH = load_a_frag(hA, F, kt, lane);
#pragma unroll
      for (int g = 0; g < 4; ++g)
#pragma unroll
        for (int j = 0; j < TW; ++j) {
          int nt = g * CPG + wave * TW + j;
          acc[g][j] = wmma_bf16(aX, load_b_frag(pWih, KT, nt, kt, lane), acc[g][j]);
          acc[g][j] = wmma_bf16(aH, load_b_frag(pWhh, KT, nt, kt, lane), acc[g][j]);
        }
    }

    // ---- phase C: register-resident LSTM cell update (masked) ----------
#pragma unroll
    for (int j = 0; j < TW; ++j)
#pragma unroll
      for (int r = 0; r < 8; ++r) {
        float gi = acc[0][j][r] + bI[j];
        float gf = acc[1][j][r] + bF[j];
        float gg = acc[2][j][r] + bG[j];
        float go = acc[3][j][r] + bO[j];
        float cn = sigm(gf) * cReg[j][r] + sigm(gi) * tanhf(gg);
        float hn = sigm(go) * tanhf(cn);
        if (t < degR[r]) { cReg[j][r] = cn; hReg[j][r] = hn; }
      }
    __syncthreads();   // all waves done reading xA/hA for this step
  }

  // ---- epilogue: out = x @ Wself^T + h @ Wneigh^T + b ------------------
  // Publish final h; stage this layer's own-node input rows into xA.
#pragma unroll
  for (int j = 0; j < TW; ++j)
#pragma unroll
    for (int r = 0; r < 8; ++r)
      hA[(m0 + r) * F + colBase + j * 16] = f2bf(hReg[j][r]);

  if (FIRST) {
#pragma unroll
    for (int it = 0; it < 2; ++it) {
      int e = tid + it * 256;
      int r = e / (F / 4), c4 = e % (F / 4);
      const float4 v = *(const float4*)(xF32 + (size_t)(nodeBase + r) * F + c4 * 4);
      u32x2 p; p.x = pack2bf(v.x, v.y); p.y = pack2bf(v.z, v.w);
      *(u32x2*)(xA + r * F + c4 * 4) = p;
    }
  } else {
#pragma unroll
    for (int it = 0; it < 2; ++it) {
      int e = tid + it * 256;
      int r = e / (F / 8), c8 = e % (F / 8);
      *(u32x4*)(xA + r * F + c8 * 8) =
          *(const u32x4*)(xBF + (size_t)(nodeBase + r) * F + c8 * 8);
    }
  }
  __syncthreads();

  for (int nt = wave; nt < OUTF / 16; nt += 8) {
    v8f acc;
#pragma unroll
    for (int r = 0; r < 8; ++r) acc[r] = 0.0f;
#pragma unroll
    for (int kt = 0; kt < KT; ++kt) {
      v16bf aS = load_a_frag(xA, F, kt, lane);
      acc = wmma_bf16(aS, load_b_frag(pWself, KT, nt, kt, lane), acc);
      v16bf aN = load_a_frag(hA, F, kt, lane);
      acc = wmma_bf16(aN, load_b_frag(pWneigh, KT, nt, kt, lane), acc);
    }
    int nc = nt * 16 + (lane & 15);
#pragma unroll
    for (int r = 0; r < 8; ++r) {
      float v = acc[r] + bout[nc];
      if (FIRST) {
        v = v > 0.0f ? v : 0.0f;
        outBF[(size_t)(nodeBase + m0 + r) * OUTF + nc] = f2bf(v);
      } else {
        outF32[(size_t)(nodeBase + m0 + r) * OUTF + nc] = v;
      }
    }
  }
}

// ---------------------------------------------------------------------------
// Host launcher
// ---------------------------------------------------------------------------
extern "C" void kernel_launch(void* const* d_in, const int* in_sizes, int n_in,
                              void* d_out, int out_size, void* d_ws, size_t ws_size,
                              hipStream_t stream) {
  (void)in_sizes; (void)n_in; (void)out_size; (void)ws_size;
  constexpr int N = 30000, IN = 128, HID = 256, OUT = 64;

  const float* feat    = (const float*)d_in[0];
  const int*   nbr     = (const int*)  d_in[1];
  const int*   deg     = (const int*)  d_in[2];
  const float* Wih1    = (const float*)d_in[3];
  const float* Whh1    = (const float*)d_in[4];
  const float* bih1    = (const float*)d_in[5];
  const float* bhh1    = (const float*)d_in[6];
  const float* Wself1  = (const float*)d_in[7];
  const float* Wneigh1 = (const float*)d_in[8];
  const float* b1      = (const float*)d_in[9];
  const float* Wih2    = (const float*)d_in[10];
  const float* Whh2    = (const float*)d_in[11];
  const float* bih2    = (const float*)d_in[12];
  const float* bhh2    = (const float*)d_in[13];
  const float* Wself2  = (const float*)d_in[14];
  const float* Wneigh2 = (const float*)d_in[15];
  const float* b2      = (const float*)d_in[16];

  // Workspace layout (bf16 element offsets)
  __bf16* ws = (__bf16*)d_ws;
  __bf16* pWih1    = ws;                       // 512*128
  __bf16* pWhh1    = pWih1    + 512 * 128;
  __bf16* pWself1  = pWhh1    + 512 * 128;     // 256*128
  __bf16* pWneigh1 = pWself1  + 256 * 128;
  __bf16* pWih2    = pWneigh1 + 256 * 128;     // 1024*256
  __bf16* pWhh2    = pWih2    + 1024 * 256;
  __bf16* pWself2  = pWhh2    + 1024 * 256;    // 64*256
  __bf16* pWneigh2 = pWself2  + 64 * 256;
  __bf16* h1       = pWneigh2 + 64 * 256;      // N*HID bf16

  auto pk = [&](const float* src, __bf16* dst, int R, int C) {
    int total = (R / 16) * (C / 32) * 32;
    pack_weights<<<(total + 255) / 256, 256, 0, stream>>>(src, dst, R, C);
  };
  pk(Wih1,    pWih1,    4 * IN,  IN);
  pk(Whh1,    pWhh1,    4 * IN,  IN);
  pk(Wself1,  pWself1,  HID,     IN);
  pk(Wneigh1, pWneigh1, HID,     IN);
  pk(Wih2,    pWih2,    4 * HID, HID);
  pk(Whh2,    pWhh2,    4 * HID, HID);
  pk(Wself2,  pWself2,  OUT,     HID);
  pk(Wneigh2, pWneigh2, OUT,     HID);

  sage_layer<128, 256, true><<<N / 16, 256, 0, stream>>>(
      feat, nullptr, nbr, deg, bih1, bhh1, b1,
      pWih1, pWhh1, pWself1, pWneigh1, h1, nullptr);

  sage_layer<256, 64, false><<<N / 16, 256, 0, stream>>>(
      nullptr, h1, nbr, deg, bih2, bhh2, b2,
      pWih2, pWhh2, pWself2, pWneigh2, nullptr, (float*)d_out);
}
